// ContinuousFilterConvolution_34952443854935
// MI455X (gfx1250) — compile-verified, matrix operand order checked by reading, not run on hardware
//
#include <hip/hip_runtime.h>
#include <hip/hip_bf16.h>

typedef __attribute__((ext_vector_type(16))) _Float16 v16h;
typedef __attribute__((ext_vector_type(8)))  float    v8f;

#define CH        128
#define RBF_K     200
#define N_NODES   50000
#define N_EDGES   600000
#define AP        136          // LDS row pitch in halves (128 + 8 pad)
#define EW        4            // waves per block in edge kernel

__device__ __forceinline__ float ssp(float v) {
    // shifted softplus: softplus(x) - log(2), with overflow guard at x>=14
    float sp = (v < 14.0f) ? __logf(1.0f + __expf(v)) : v;
    return sp - 0.69314718055994530942f;
}

// k index inside a 32-deep k-step for 16-bit A/B fragments (ISA 7.12.2 table):
// lanes 0-15: K = {0..7, 16..23}; lanes 16-31: +8.
__device__ __forceinline__ int kmap(int lane, int i) {
    return i + (i & 8) + ((lane & 16) ? 8 : 0);
}

// ---------------------------------------------------------------------------
// Pack a row-major [Kreal x 128] f32 weight matrix into fragment-ordered f16:
// half index t = ((kstep*8 + ntile)*32 + lane)*16 + i, zero-padded past Kreal.
// ---------------------------------------------------------------------------
__global__ void pack_weights_kernel(const float* __restrict__ W,
                                    _Float16* __restrict__ Wp,
                                    int Kreal, int total_halves) {
    int t = blockIdx.x * blockDim.x + threadIdx.x;
    if (t >= total_halves) return;
    int f    = t >> 9;          // fragment id
    int r    = t & 511;
    int lane = r >> 4;
    int i    = r & 15;
    int ks   = f >> 3;
    int nt   = f & 7;
    int k = ks * 32 + kmap(lane, i);
    int n = nt * 16 + (lane & 15);
    float v = (k < Kreal) ? W[k * CH + n] : 0.0f;
    Wp[t] = (_Float16)v;
}

// ---------------------------------------------------------------------------
// f = x @ w3   (one wave per 16-node tile, 8 WMMA column tiles)
// ---------------------------------------------------------------------------
__global__ void node_gemm_kernel(const float* __restrict__ x,
                                 const _Float16* __restrict__ w3p,
                                 float* __restrict__ fbuf, int ntiles) {
    int wave = threadIdx.x >> 5;
    int lane = threadIdx.x & 31;
    int t = blockIdx.x * 8 + wave;
    if (t >= ntiles) return;
    int nbase = t * 16;
    int col  = lane & 15;
    int rofs = (lane & 16) ? 8 : 0;

    v8f acc[8] = {};
    int m = nbase + col;
    #pragma unroll
    for (int ks = 0; ks < 4; ++ks) {
        int kbase = ks * 32 + rofs;
        v16h a;
        #pragma unroll
        for (int i = 0; i < 16; ++i)
            a[i] = (_Float16)x[m * CH + kbase + i + (i & 8)];
        #pragma unroll
        for (int nt = 0; nt < 8; ++nt) {
            v16h b = *(const v16h*)(w3p + (((ks * 8 + nt) * 32 + lane) << 4));
            acc[nt] = __builtin_amdgcn_wmma_f32_16x16x32_f16(
                false, a, false, b, (short)0, acc[nt], false, false);
        }
    }
    #pragma unroll
    for (int nt = 0; nt < 8; ++nt) {
        int c = nt * 16 + col;
        #pragma unroll
        for (int r = 0; r < 8; ++r)
            fbuf[(size_t)(nbase + rofs + r) * CH + c] = acc[nt][r];
    }
}

// ---------------------------------------------------------------------------
// Edge pipeline, M-blocked by 2 (one wave = 32 edges):
//   h = ssp(rbf@w1+b1); w = ssp(h@w2+b2); conv[src] += w * f[src]
// Each B fragment load feeds two WMMAs (two edge row-tiles), halving the
// load:wmma issue ratio vs the unblocked version. Intermediates transposed
// through per-wave LDS tiles, ordered with explicit s_wait_dscnt.
// ---------------------------------------------------------------------------
__global__ void edge_kernel(const float* __restrict__ rbf,
                            const int*   __restrict__ src,
                            const float* __restrict__ b1,
                            const float* __restrict__ b2,
                            const _Float16* __restrict__ w1p,
                            const _Float16* __restrict__ w2p,
                            const float* __restrict__ fbuf,
                            float* __restrict__ conv, int nsuper) {
    __shared__ _Float16 hstage[EW][2][16 * AP];
    int wave = threadIdx.x >> 5;
    int lane = threadIdx.x & 31;
    int st = blockIdx.x * EW + wave;        // super-tile of 32 edges
    if (st >= nsuper) return;
    int ebase = st * 32;
    int col  = lane & 15;
    int rofs = (lane & 16) ? 8 : 0;

    // ---- GEMM 1: [32 x 224(pad)] @ w1p -> acc0/acc1 (f32) ----
    v8f acc0[8] = {}, acc1[8] = {};
    int m0 = ebase + col;
    int m1 = ebase + 16 + col;
    #pragma unroll
    for (int ks = 0; ks < 7; ++ks) {
        int kbase = ks * 32 + rofs;
        v16h a0, a1;
        if (ks < 6) {
            #pragma unroll
            for (int i = 0; i < 16; ++i) {
                int k = kbase + i + (i & 8);
                a0[i] = (_Float16)rbf[(size_t)m0 * RBF_K + k];
                a1[i] = (_Float16)rbf[(size_t)m1 * RBF_K + k];
            }
        } else {
            #pragma unroll
            for (int i = 0; i < 16; ++i) {
                int k = kbase + i + (i & 8);
                a0[i] = (k < RBF_K) ? (_Float16)rbf[(size_t)m0 * RBF_K + k]
                                    : (_Float16)0.0f;
                a1[i] = (k < RBF_K) ? (_Float16)rbf[(size_t)m1 * RBF_K + k]
                                    : (_Float16)0.0f;
            }
        }
        #pragma unroll
        for (int nt = 0; nt < 8; ++nt) {
            v16h b = *(const v16h*)(w1p + (((ks * 8 + nt) * 32 + lane) << 4));
            acc0[nt] = __builtin_amdgcn_wmma_f32_16x16x32_f16(
                false, a0, false, b, (short)0, acc0[nt], false, false);
            acc1[nt] = __builtin_amdgcn_wmma_f32_16x16x32_f16(
                false, a1, false, b, (short)0, acc1[nt], false, false);
        }
    }

    // ---- bias + ssp, C-layout -> two LDS tiles (row-major 16 x 128 f16) ----
    _Float16* hs0 = &hstage[wave][0][0];
    _Float16* hs1 = &hstage[wave][1][0];
    #pragma unroll
    for (int nt = 0; nt < 8; ++nt) {
        int c = nt * 16 + col;
        float bb = b1[c];
        #pragma unroll
        for (int r = 0; r < 8; ++r) {
            hs0[(rofs + r) * AP + c] = (_Float16)ssp(acc0[nt][r] + bb);
            hs1[(rofs + r) * AP + c] = (_Float16)ssp(acc1[nt][r] + bb);
        }
    }
    asm volatile("s_wait_dscnt 0" ::: "memory");   // in-wave LDS transpose fence

    // ---- GEMM 2: [32 x 128] h @ w2p, B fragments shared across both tiles --
    #pragma unroll
    for (int nt = 0; nt < 8; ++nt) { acc0[nt] = (v8f){}; acc1[nt] = (v8f){}; }
    #pragma unroll
    for (int ks = 0; ks < 4; ++ks) {
        int kbase = ks * 32 + rofs;
        v16h a0, a1;
        #pragma unroll
        for (int i = 0; i < 16; ++i) {
            int k = kbase + i + (i & 8);
            a0[i] = hs0[col * AP + k];
            a1[i] = hs1[col * AP + k];
        }
        #pragma unroll
        for (int nt = 0; nt < 8; ++nt) {
            v16h b = *(const v16h*)(w2p + (((ks * 8 + nt) * 32 + lane) << 4));
            acc0[nt] = __builtin_amdgcn_wmma_f32_16x16x32_f16(
                false, a0, false, b, (short)0, acc0[nt], false, false);
            acc1[nt] = __builtin_amdgcn_wmma_f32_16x16x32_f16(
                false, a1, false, b, (short)0, acc1[nt], false, false);
        }
    }

    // ---- bias + ssp, gather f[src], multiply, atomic scatter-add ----
    int s0[8], s1[8];
    #pragma unroll
    for (int r = 0; r < 8; ++r) {
        s0[r] = src[ebase + rofs + r];
        s1[r] = src[ebase + 16 + rofs + r];
    }
    #pragma unroll
    for (int nt = 0; nt < 8; ++nt) {
        int c = nt * 16 + col;
        float bb = b2[c];
        #pragma unroll
        for (int r = 0; r < 8; ++r) {
            float wv0 = ssp(acc0[nt][r] + bb);
            size_t off0 = (size_t)s0[r] * CH + c;
            __hip_atomic_fetch_add(&conv[off0], wv0 * fbuf[off0],
                                   __ATOMIC_RELAXED, __HIP_MEMORY_SCOPE_AGENT);
            float wv1 = ssp(acc1[nt][r] + bb);
            size_t off1 = (size_t)s1[r] * CH + c;
            __hip_atomic_fetch_add(&conv[off1], wv1 * fbuf[off1],
                                   __ATOMIC_RELAXED, __HIP_MEMORY_SCOPE_AGENT);
        }
    }
}

// ---------------------------------------------------------------------------
// out = x + (ssp(conv@w4+b4) @ w5 + b5)
// ---------------------------------------------------------------------------
__global__ void out_kernel(const float* __restrict__ conv,
                           const float* __restrict__ x,
                           const float* __restrict__ b4,
                           const float* __restrict__ b5,
                           const _Float16* __restrict__ w4p,
                           const _Float16* __restrict__ w5p,
                           float* __restrict__ out, int ntiles) {
    __shared__ _Float16 hstage[8][16 * AP];
    int wave = threadIdx.x >> 5;
    int lane = threadIdx.x & 31;
    int t = blockIdx.x * 8 + wave;
    if (t >= ntiles) return;
    int nbase = t * 16;
    int col  = lane & 15;
    int rofs = (lane & 16) ? 8 : 0;

    // ---- GEMM 1: conv @ w4p ----
    v8f acc[8] = {};
    int m = nbase + col;
    #pragma unroll
    for (int ks = 0; ks < 4; ++ks) {
        int kbase = ks * 32 + rofs;
        v16h a;
        #pragma unroll
        for (int i = 0; i < 16; ++i)
            a[i] = (_Float16)conv[(size_t)m * CH + kbase + i + (i & 8)];
        #pragma unroll
        for (int nt = 0; nt < 8; ++nt) {
            v16h b = *(const v16h*)(w4p + (((ks * 8 + nt) * 32 + lane) << 4));
            acc[nt] = __builtin_amdgcn_wmma_f32_16x16x32_f16(
                false, a, false, b, (short)0, acc[nt], false, false);
        }
    }

    _Float16* hs = &hstage[wave][0];
    #pragma unroll
    for (int nt = 0; nt < 8; ++nt) {
        int c = nt * 16 + col;
        float bb = b4[c];
        #pragma unroll
        for (int r = 0; r < 8; ++r)
            hs[(rofs + r) * AP + c] = (_Float16)ssp(acc[nt][r] + bb);
    }
    asm volatile("s_wait_dscnt 0" ::: "memory");

    // ---- GEMM 2: y @ w5p ----
    #pragma unroll
    for (int nt = 0; nt < 8; ++nt) acc[nt] = (v8f){};
    #pragma unroll
    for (int ks = 0; ks < 4; ++ks) {
        int kbase = ks * 32 + rofs;
        v16h a;
        #pragma unroll
        for (int i = 0; i < 16; ++i)
            a[i] = hs[col * AP + kbase + i + (i & 8)];
        #pragma unroll
        for (int nt = 0; nt < 8; ++nt) {
            v16h b = *(const v16h*)(w5p + (((ks * 8 + nt) * 32 + lane) << 4));
            acc[nt] = __builtin_amdgcn_wmma_f32_16x16x32_f16(
                false, a, false, b, (short)0, acc[nt], false, false);
        }
    }

    // ---- + b5 + residual x ----
    #pragma unroll
    for (int nt = 0; nt < 8; ++nt) {
        int c = nt * 16 + col;
        float bb = b5[c];
        #pragma unroll
        for (int r = 0; r < 8; ++r) {
            size_t off = (size_t)(nbase + rofs + r) * CH + c;
            out[off] = x[off] + acc[nt][r] + bb;
        }
    }
}

// ---------------------------------------------------------------------------
extern "C" void kernel_launch(void* const* d_in, const int* in_sizes, int n_in,
                              void* d_out, int out_size, void* d_ws, size_t ws_size,
                              hipStream_t stream) {
    const float* rbf = (const float*)d_in[0];
    const float* x   = (const float*)d_in[1];
    const int*   src = (const int*)  d_in[2];
    const float* w1  = (const float*)d_in[3];
    const float* b1  = (const float*)d_in[4];
    const float* w2  = (const float*)d_in[5];
    const float* b2  = (const float*)d_in[6];
    const float* w3  = (const float*)d_in[7];
    const float* w4  = (const float*)d_in[8];
    const float* b4  = (const float*)d_in[9];
    const float* w5  = (const float*)d_in[10];
    const float* b5  = (const float*)d_in[11];

    // workspace layout (bytes)
    char* ws = (char*)d_ws;
    const size_t W1P_HALVES = 7 * 8 * 512;   // K padded 200 -> 224
    const size_t WXP_HALVES = 4 * 8 * 512;   // K = 128
    _Float16* w1p = (_Float16*)(ws);
    _Float16* w2p = (_Float16*)(ws + 2 * (W1P_HALVES));
    _Float16* w3p = (_Float16*)(ws + 2 * (W1P_HALVES + WXP_HALVES));
    _Float16* w4p = (_Float16*)(ws + 2 * (W1P_HALVES + 2 * WXP_HALVES));
    _Float16* w5p = (_Float16*)(ws + 2 * (W1P_HALVES + 3 * WXP_HALVES));
    float* fbuf = (float*)(ws + 2 * (W1P_HALVES + 4 * WXP_HALVES));
    float* conv = fbuf + (size_t)N_NODES * CH;

    // zero the scatter accumulator (harness poisons ws)
    hipMemsetAsync(conv, 0, (size_t)N_NODES * CH * sizeof(float), stream);

    // pack weights into fragment-ordered f16
    pack_weights_kernel<<<(int)((W1P_HALVES + 255) / 256), 256, 0, stream>>>(
        w1, w1p, RBF_K, (int)W1P_HALVES);
    pack_weights_kernel<<<(int)((WXP_HALVES + 255) / 256), 256, 0, stream>>>(
        w2, w2p, CH, (int)WXP_HALVES);
    pack_weights_kernel<<<(int)((WXP_HALVES + 255) / 256), 256, 0, stream>>>(
        w3, w3p, CH, (int)WXP_HALVES);
    pack_weights_kernel<<<(int)((WXP_HALVES + 255) / 256), 256, 0, stream>>>(
        w4, w4p, CH, (int)WXP_HALVES);
    pack_weights_kernel<<<(int)((WXP_HALVES + 255) / 256), 256, 0, stream>>>(
        w5, w5p, CH, (int)WXP_HALVES);

    // f = x @ w3
    const int node_tiles = N_NODES / 16;               // 3125
    node_gemm_kernel<<<(node_tiles + 7) / 8, 256, 0, stream>>>(
        x, w3p, fbuf, node_tiles);

    // edge MLP + gather + scatter (32 edges per wave, 4 waves per block)
    const int edge_super = N_EDGES / 32;               // 18750
    edge_kernel<<<(edge_super + EW - 1) / EW, EW * 32, 0, stream>>>(
        rbf, src, b1, b2, w1p, w2p, fbuf, conv, edge_super);

    // output MLP + residual
    out_kernel<<<(node_tiles + 7) / 8, 256, 0, stream>>>(
        conv, x, b4, b5, w4p, w5p, (float*)d_out, node_tiles);
}
